// CausalSelfAttentionMarcin_31464930411196
// MI455X (gfx1250) — matrix operand order repaired and checked by reading, compiled, not run on hardware
//
#include <hip/hip_runtime.h>

// ---------------------------------------------------------------------------
// CDNA5 (gfx1250) causal self-attention.
// Prologues (cvt x->bf16, transpose W->bf16^T, memory-bound, once)
//   -> QKV GEMM (WMMA bf16, async-to-LDS double-buffered)
//   -> flash attention (WMMA bf16, async-to-LDS K tiles)
//   -> proj GEMM.
// Matrix math: v_wmma_f32_16x16x32_bf16. Staging: global_load_async_to_lds.
// ---------------------------------------------------------------------------

typedef __attribute__((ext_vector_type(16))) __bf16   v16bf;
typedef __attribute__((ext_vector_type(8)))  float    v8f;
typedef __attribute__((ext_vector_type(4)))  unsigned u32x4;
typedef __attribute__((ext_vector_type(4)))  float    f32x4;

#define BATCH 4
#define TSEQ  2048
#define CDIM  1024
#define NHEAD 16
#define HDIM  64

union FragU { v16bf f; u32x4 q[2]; };

__device__ __forceinline__ unsigned short bfbits(float f) {
  unsigned u = __builtin_bit_cast(unsigned, f);
  u = (u + 0x7FFFu + ((u >> 16) & 1u)) >> 16;   // round-to-nearest-even
  return (unsigned short)u;
}
__device__ __forceinline__ __bf16 f2bf(float f) {
  unsigned short s = bfbits(f);
  return __builtin_bit_cast(__bf16, s);
}
__device__ __forceinline__ unsigned pk2bf(float lo, float hi) {
  return (unsigned)bfbits(lo) | ((unsigned)bfbits(hi) << 16);
}
// Build a v16bf WMMA fragment from two 16-byte runs (b128 loads).
__device__ __forceinline__ v16bf load_frag2(const __bf16* p0, const __bf16* p1) {
  FragU u;
  u.q[0] = *(const u32x4*)p0;
  u.q[1] = *(const u32x4*)p1;
  return u.f;
}

// Flat shared-aperture address truncates to the 0-based LDS byte offset
// (see ISA 10.2).
__device__ __forceinline__ unsigned lds_off(const void* p) {
  return (unsigned)(size_t)p;
}
// GLOBAL_LOAD_ASYNC_TO_LDS_B128, GVS mode: saddr(SGPR64) + vaddr(VGPR i32).
// Tracked by ASYNCcnt; LDS written directly (no VGPR round-trip).
__device__ __forceinline__ void async_b128(unsigned ldsoff, unsigned voff,
                                           const void* sbase) {
  asm volatile("global_load_async_to_lds_b128 %0, %1, %2"
               :
               : "v"(ldsoff), "v"(voff), "s"(sbase)
               : "memory");
}
__device__ __forceinline__ void wait_async0() {
  asm volatile("s_wait_asynccnt 0x0" ::: "memory");
}

// ---------------------------------------------------------------------------
// Prologue 1: elementwise fp32 -> bf16 (8 elements / thread, b128 in/out)
// ---------------------------------------------------------------------------
__global__ __launch_bounds__(256) void cvt_f32_bf16(
    const float* __restrict__ src, __bf16* __restrict__ dst, long n) {
  const long i = ((long)blockIdx.x * 256 + threadIdx.x) * 8;
  if (i + 8 <= n) {
    f32x4 f0 = *(const f32x4*)(src + i);
    f32x4 f1 = *(const f32x4*)(src + i + 4);
    u32x4 q = {pk2bf(f0[0], f0[1]), pk2bf(f0[2], f0[3]),
               pk2bf(f1[0], f1[1]), pk2bf(f1[2], f1[3])};
    *(u32x4*)(dst + i) = q;
  }
}

// ---------------------------------------------------------------------------
// Prologue 2: W[K][N] fp32 -> Wt[N][K] bf16 (32x32 LDS tile transpose)
// ---------------------------------------------------------------------------
__global__ __launch_bounds__(256) void transpose_w_bf16(
    const float* __restrict__ W, __bf16* __restrict__ Wt, int K, int N) {
  __shared__ float tile[32][33];
  const int k0 = blockIdx.x * 32;
  const int n0 = blockIdx.y * 32;
  const int tr = threadIdx.x >> 5;     // 0..7
  const int tc = threadIdx.x & 31;
#pragma unroll
  for (int i = 0; i < 4; ++i)
    tile[tr + i * 8][tc] = W[(size_t)(k0 + tr + i * 8) * N + n0 + tc];
  __syncthreads();
#pragma unroll
  for (int i = 0; i < 4; ++i)
    Wt[(size_t)(n0 + tr + i * 8) * K + k0 + tc] = f2bf(tile[tc][tr + i * 8]);
}

// ---------------------------------------------------------------------------
// GEMM: Out[M,N] = A[M,K] * Wt[N,K]^T + bias[N]   (A, Wt bf16)
// Block tile 64x128, 128 threads (4 waves as 2x2); wave tile 32x64 = 2x4 WMMA.
// Double-buffered LDS; tiles staged with global_load_async_to_lds_b128 issued
// one iteration ahead, synchronized with s_wait_asynccnt + barrier.
// ---------------------------------------------------------------------------
template <bool OUT_BF16>
__global__ __launch_bounds__(128) void gemm_bf16(
    const __bf16* __restrict__ A, const __bf16* __restrict__ Wt,
    const float* __restrict__ bias, void* __restrict__ Out,
    int M, int N, int K) {
  __shared__ __bf16 ldsA[2][64][40];     // [buf][m][k]
  __shared__ __bf16 ldsBt[2][128][40];   // [buf][n][k]

  const int tid  = threadIdx.x;
  const int lane = tid & 31;
  const int wave = tid >> 5;
  const int wm   = wave >> 1;            // 0..1 -> 32 rows
  const int wn   = wave & 1;             // 0..1 -> 64 cols
  const int hi   = (lane >> 4) & 1;
  const int ln   = lane & 15;

  const int bm = blockIdx.x * 64;
  const int bn = blockIdx.y * 128;

  // Per-thread staging addresses (byte offsets, GVS mode).
  const int rA = tid >> 1;
  const int cA = (tid & 1) * 16;
  const unsigned gAoff = (unsigned)((unsigned)(bm + rA) * (unsigned)K + cA) * 2u;
  const unsigned gBoff = (unsigned)((unsigned)(bn + tid) * (unsigned)K) * 2u;
  unsigned lAoff[2], lBoff[2];
#pragma unroll
  for (int i = 0; i < 2; ++i) {
    lAoff[i] = lds_off(&ldsA[i][rA][cA]);
    lBoff[i] = lds_off(&ldsBt[i][tid][0]);
  }

  auto stage = [&](int buf, int k0) {
    const unsigned va = gAoff + (unsigned)k0 * 2u;
    async_b128(lAoff[buf],       va,       A);
    async_b128(lAoff[buf] + 16,  va + 16,  A);
    const unsigned vb = gBoff + (unsigned)k0 * 2u;
    async_b128(lBoff[buf],       vb,       Wt);
    async_b128(lBoff[buf] + 16,  vb + 16,  Wt);
    async_b128(lBoff[buf] + 32,  vb + 32,  Wt);
    async_b128(lBoff[buf] + 48,  vb + 48,  Wt);
  };

  v8f c[2][4] = {};

  stage(0, 0);
  wait_async0();
  __syncthreads();

  for (int k0 = 0; k0 < K; k0 += 32) {
    const int cur = (k0 >> 5) & 1;
    if (k0 + 32 < K) stage(cur ^ 1, k0 + 32);   // prefetch next tile (async)

    // ---- fragments: contiguous b128 runs per ISA layouts ----
    v16bf a[2], b[4];
#pragma unroll
    for (int mt = 0; mt < 2; ++mt) {
      const __bf16* ap = &ldsA[cur][wm * 32 + mt * 16 + ln][hi * 8];
      a[mt] = load_frag2(ap, ap + 16);
    }
#pragma unroll
    for (int nt = 0; nt < 4; ++nt) {
      const __bf16* bp = &ldsBt[cur][wn * 64 + nt * 16 + ln][hi * 8];
      b[nt] = load_frag2(bp, bp + 16);
    }

#pragma unroll
    for (int mt = 0; mt < 2; ++mt)
#pragma unroll
      for (int nt = 0; nt < 4; ++nt)
        c[mt][nt] = __builtin_amdgcn_wmma_f32_16x16x32_bf16(
            false, a[mt], false, b[nt], (short)0, c[mt][nt], false, false);

    wait_async0();     // own async loads (next buffer) complete
    __syncthreads();   // all waves done -> next buffer valid, cur reusable
  }

  // ---- epilogue: bias add + store per f32 C/D layout ----
#pragma unroll
  for (int mt = 0; mt < 2; ++mt) {
#pragma unroll
    for (int nt = 0; nt < 4; ++nt) {
      const int n = bn + wn * 64 + nt * 16 + ln;
      const float bv = bias[n];
#pragma unroll
      for (int r = 0; r < 8; ++r) {
        const int m = bm + wm * 32 + mt * 16 + r + (hi ? 8 : 0);
        const float val = c[mt][nt][r] + bv;
        if (OUT_BF16)
          ((__bf16*)Out)[(size_t)m * N + n] = f2bf(val);
        else
          ((float*)Out)[(size_t)m * N + n] = val;
      }
    }
  }
}

// ---------------------------------------------------------------------------
// Flash attention: block = (b,h, 64-row q tile), 4 waves x 16 q rows each.
// qkv bf16 [B,T,3C]; y bf16 [B,T,C].
// ldsK [key][d] staged via async-to-LDS (B-frag order for Q*K^T);
// ldsVt [d][key] staged via VGPR transpose (B-frag order for P*V).
// ---------------------------------------------------------------------------
__global__ __launch_bounds__(128) void attn_kernel(
    const __bf16* __restrict__ qkv, __bf16* __restrict__ y) {
  __shared__ __bf16 ldsK[64][72];
  __shared__ __bf16 ldsVt[64][72];
  __shared__ __bf16 ldsP[4][16][72];   // per-wave P re-layout buffer [m][key]

  const int tid  = threadIdx.x;
  const int lane = tid & 31;
  const int wave = tid >> 5;
  const int hi   = (lane >> 4) & 1;
  const int ln   = lane & 15;

  const int bh = blockIdx.x;
  const int b  = bh >> 4;
  const int h  = bh & (NHEAD - 1);
  const int qb = blockIdx.y;
  const int q0 = qb * 64;

  const unsigned rs = 3 * CDIM;
  const __bf16* qbase = qkv + (size_t)b * TSEQ * rs + h * HDIM;
  const __bf16* kbase = qbase + CDIM;
  const __bf16* vbase = qbase + 2 * CDIM;

  // Q fragments (A-layout): two contiguous 16B runs per lane -> b128 loads
  v16bf aQ[2];
  {
    const __bf16* qp = qbase + (size_t)(q0 + wave * 16 + ln) * rs;
#pragma unroll
    for (int ks = 0; ks < 2; ++ks)
      aQ[ks] = load_frag2(qp + ks * 32 + hi * 8, qp + ks * 32 + hi * 8 + 16);
  }

  // Per-thread staging addresses.
  const int rK  = tid >> 1;
  const int cK  = (tid & 1) * 32;
  const unsigned lKoff = lds_off(&ldsK[rK][cK]);

  v8f   o[4] = {};
  float rmax[8], rsum[8];
#pragma unroll
  for (int r = 0; r < 8; ++r) { rmax[r] = -1e30f; rsum[r] = 0.0f; }

  for (int kt = 0; kt <= qb; ++kt) {
    // ---- stage K tile via async-to-LDS (4x b128, no VGPR round-trip) ----
    {
      const unsigned voff = ((unsigned)(kt * 64 + rK) * rs + cK) * 2u;
      async_b128(lKoff,      voff,      kbase);
      async_b128(lKoff + 16, voff + 16, kbase);
      async_b128(lKoff + 32, voff + 32, kbase);
      async_b128(lKoff + 48, voff + 48, kbase);
    }
    // ---- stage V tile transposed [d][key], packing key-pairs per b32 ----
    {
      const int kp = tid & 31;            // keys 2kp, 2kp+1
      const int d0 = (tid >> 5) * 16;     // 16 d-values
      const __bf16* s0 = vbase + (size_t)(kt * 64 + 2 * kp) * rs + d0;
      const __bf16* s1 = s0 + rs;
#pragma unroll
      for (int half = 0; half < 2; ++half) {
        u32x4 r0 = ((const u32x4*)s0)[half];
        u32x4 r1 = ((const u32x4*)s1)[half];
#pragma unroll
        for (int w = 0; w < 4; ++w) {
          const int d = d0 + half * 8 + 2 * w;
          *(unsigned*)&ldsVt[d][2 * kp]     = (r0[w] & 0xFFFFu) | (r1[w] << 16);
          *(unsigned*)&ldsVt[d + 1][2 * kp] = (r0[w] >> 16) | (r1[w] & 0xFFFF0000u);
        }
      }
      if (kt < qb) __builtin_prefetch(s0 + 64 * (size_t)rs, 0, 0);
    }
    wait_async0();
    __syncthreads();

    // ---- S = Q * K^T : four 16x16 tiles along the 64 keys ----
    v8f s[4];
#pragma unroll
    for (int j = 0; j < 4; ++j) {
      v8f acc = {};
#pragma unroll
      for (int ks = 0; ks < 2; ++ks) {
        const __bf16* kpp = &ldsK[j * 16 + ln][ks * 32 + hi * 8];
        v16bf bK = load_frag2(kpp, kpp + 16);
        acc = __builtin_amdgcn_wmma_f32_16x16x32_bf16(
            false, aQ[ks], false, bK, (short)0, acc, false, false);
      }
      s[j] = acc;
    }

    // ---- scale + causal mask (diagonal tile only) ----
    const bool diag = (kt == qb);
#pragma unroll
    for (int j = 0; j < 4; ++j)
#pragma unroll
      for (int r = 0; r < 8; ++r) {
        float v = s[j][r] * 0.125f;       // 1/sqrt(64)
        if (diag) {
          const int m = wave * 16 + r + hi * 8;
          const int n = j * 16 + ln;
          if (n > m) v = -1e30f;
        }
        s[j][r] = v;
      }

    // ---- online softmax (row stats via 16-lane-group shuffles) ----
    float alpha[8];
#pragma unroll
    for (int r = 0; r < 8; ++r) {
      float mx = -1e30f;
#pragma unroll
      for (int j = 0; j < 4; ++j) mx = fmaxf(mx, s[j][r]);
      mx = fmaxf(mx, __shfl_xor(mx, 8, 32));
      mx = fmaxf(mx, __shfl_xor(mx, 4, 32));
      mx = fmaxf(mx, __shfl_xor(mx, 2, 32));
      mx = fmaxf(mx, __shfl_xor(mx, 1, 32));
      const float mnew = fmaxf(rmax[r], mx);
      alpha[r] = __builtin_exp2f((rmax[r] - mnew) * 1.44269504f);
      float rsloc = 0.0f;
#pragma unroll
      for (int j = 0; j < 4; ++j) {
        const float p = __builtin_exp2f((s[j][r] - mnew) * 1.44269504f);
        s[j][r] = p;
        rsloc += p;
      }
      rsloc += __shfl_xor(rsloc, 8, 32);
      rsloc += __shfl_xor(rsloc, 4, 32);
      rsloc += __shfl_xor(rsloc, 2, 32);
      rsloc += __shfl_xor(rsloc, 1, 32);
      rsum[r] = rsum[r] * alpha[r] + rsloc;
      rmax[r] = mnew;
    }
#pragma unroll
    for (int j = 0; j < 4; ++j)
#pragma unroll
      for (int r = 0; r < 8; ++r) o[j][r] *= alpha[r];

    // ---- re-layout P (C-format -> A-format) via wave-private LDS ----
#pragma unroll
    for (int j = 0; j < 4; ++j)
#pragma unroll
      for (int r = 0; r < 8; ++r)
        ldsP[wave][r + hi * 8][j * 16 + ln] = f2bf(s[j][r]);
    // wave-local RAW through LDS; compiler inserts s_wait_dscnt

    v16bf aP[2];
#pragma unroll
    for (int ks = 0; ks < 2; ++ks) {
      const __bf16* pp = &ldsP[wave][ln][ks * 32 + hi * 8];
      aP[ks] = load_frag2(pp, pp + 16);
    }

    // ---- O += P * V ----
#pragma unroll
    for (int jd = 0; jd < 4; ++jd) {
#pragma unroll
      for (int ks = 0; ks < 2; ++ks) {
        const __bf16* vp = &ldsVt[jd * 16 + ln][ks * 32 + hi * 8];
        v16bf bV = load_frag2(vp, vp + 16);
        o[jd] = __builtin_amdgcn_wmma_f32_16x16x32_bf16(
            false, aP[ks], false, bV, (short)0, o[jd], false, false);
      }
    }
    __syncthreads();   // protect ldsK/ldsVt before next tile load
  }

  // ---- normalize and store y (bf16) ----
#pragma unroll
  for (int r = 0; r < 8; ++r) rsum[r] = 1.0f / rsum[r];
#pragma unroll
  for (int jd = 0; jd < 4; ++jd)
#pragma unroll
    for (int r = 0; r < 8; ++r) {
      const int m = q0 + wave * 16 + r + hi * 8;
      const int n = h * HDIM + jd * 16 + ln;
      y[(size_t)(b * TSEQ + m) * CDIM + n] = f2bf(o[jd][r] * rsum[r]);
    }
}

// ---------------------------------------------------------------------------
// Launch: prologues -> qkv GEMM -> flash attention -> proj GEMM
// ---------------------------------------------------------------------------
extern "C" void kernel_launch(void* const* d_in, const int* in_sizes, int n_in,
                              void* d_out, int out_size, void* d_ws,
                              size_t ws_size, hipStream_t stream) {
  const float* x      = (const float*)d_in[0];
  const float* W_attn = (const float*)d_in[1];
  const float* b_attn = (const float*)d_in[2];
  const float* W_proj = (const float*)d_in[3];
  const float* b_proj = (const float*)d_in[4];
  float* out = (float*)d_out;

  const int M = BATCH * TSEQ;                  // 8192
  __bf16* qkv = (__bf16*)d_ws;                 // [M, 3C]      50.3 MB
  __bf16* yat = qkv + (size_t)M * 3 * CDIM;    // [M, C]       16.8 MB
  __bf16* xb  = yat + (size_t)M * CDIM;        // [M, C]       16.8 MB
  __bf16* WtA = xb + (size_t)M * CDIM;         // [3C, C]       6.3 MB
  __bf16* WtP = WtA + (size_t)3 * CDIM * CDIM; // [C, C]        2.1 MB

  // Prologues (memory-bound, run once): convert x, transpose+convert weights.
  {
    const long nx = (long)M * CDIM;
    cvt_f32_bf16<<<(unsigned)(nx / (256 * 8)), 256, 0, stream>>>(x, xb, nx);
    transpose_w_bf16<<<dim3(CDIM / 32, (3 * CDIM) / 32), 256, 0, stream>>>(
        W_attn, WtA, CDIM, 3 * CDIM);
    transpose_w_bf16<<<dim3(CDIM / 32, CDIM / 32), 256, 0, stream>>>(
        W_proj, WtP, CDIM, CDIM);
  }

  // 1) qkv = x @ W_attn + b_attn   (bf16 -> bf16)
  gemm_bf16<true><<<dim3(M / 64, (3 * CDIM) / 128), 128, 0, stream>>>(
      xb, WtA, b_attn, qkv, M, 3 * CDIM, CDIM);

  // 2) flash attention per (b, h, 64-row q tile)
  attn_kernel<<<dim3(BATCH * NHEAD, TSEQ / 64), 128, 0, stream>>>(qkv, yat);

  // 3) out = y @ W_proj + b_proj   (bf16 -> fp32)
  gemm_bf16<false><<<dim3(M / 64, CDIM / 128), 128, 0, stream>>>(
      yat, WtP, b_proj, out, M, CDIM, CDIM);
}
